// LatentDiffusionAlignment_11106785428025
// MI455X (gfx1250) — matrix operand order repaired and checked by reading, not compiled, not run on hardware
//
#include <hip/hip_runtime.h>
#include <hip/hip_bf16.h>
#include <math.h>

// ---------------------------------------------------------------------------
// LatentDiffusionAlignment for MI455X (gfx1250, wave32)
//   GroupNorm(fp32) -> bf16 K-pair-interleaved -> [W1 GEMM + GELU] ->
//   bf16 K-pair-interleaved -> [W2 GEMM + bias] -> residual add (fp32).
// GEMMs: v_wmma_f32_16x16x32_bf16, tiles staged by the Tensor Data Mover
// (tensor_load_to_lds, TENSORcnt) into double-buffered LDS.
// Shapes: B=2, C=512, S=D*H*W=32768, GROUPS=8.
// ---------------------------------------------------------------------------

typedef __attribute__((ext_vector_type(16))) __bf16        v16bf;
typedef __attribute__((ext_vector_type(8)))  float         v8f;
typedef __attribute__((ext_vector_type(8)))  unsigned int  v8u;
typedef __attribute__((ext_vector_type(4)))  unsigned int  u32x4v;
typedef __attribute__((ext_vector_type(8)))  int           i32x8v;
typedef __attribute__((ext_vector_type(4)))  int           i32x4v;

#define Cc    512
#define Ss    32768           // D*H*W
#define Bb    2
#define NGRP  8
#define GN_N  2097152.0f      // (C/GROUPS) * Ss
#define EPSF  1e-5f

#define SA_STRIDE 20          // dwords per A-tile row (16 used + 4 TDM pad)
#define SB_STRIDE 132         // dwords per B-tile row (128 used + 4 TDM pad)
#define SA_DW     (128 * SA_STRIDE)       // 2560 dwords
#define SB_DW     (16 * SB_STRIDE)        // 2112 dwords
#define BUF_DW    (SA_DW + SB_DW)         // 4672 dwords per pipeline stage

__device__ __forceinline__ unsigned short f2bf(float f) {
    return __builtin_bit_cast(unsigned short, (__bf16)f);
}
__device__ __forceinline__ unsigned pack2(float lo, float hi) {
    return (unsigned)f2bf(lo) | ((unsigned)f2bf(hi) << 16);
}

// ------------------------------- GroupNorm ---------------------------------

__global__ void zero_stats(float* p) {
    if (threadIdx.x < 32) p[threadIdx.x] = 0.0f;
}

// 2048 blocks: 16 (b,g) regions x 128 chunks; each region = contiguous 2M floats.
__global__ void gn_partial(const float* __restrict__ x,
                           float* __restrict__ sum, float* __restrict__ sumsq) {
    const int tid   = threadIdx.x;
    const int bg    = blockIdx.x >> 7;
    const int chunk = blockIdx.x & 127;
    const float4* x4 = (const float4*)x;
    size_t base4 = (size_t)bg * 524288 + (size_t)chunk * 4096;

    float s = 0.0f, q = 0.0f;
#pragma unroll
    for (int i = 0; i < 16; ++i) {
        float4 v = x4[base4 + (size_t)i * 256 + tid];
        s += v.x + v.y + v.z + v.w;
        q += v.x * v.x + v.y * v.y + v.z * v.z + v.w * v.w;
    }
    __shared__ float rs[256], rq[256];
    rs[tid] = s; rq[tid] = q;
    __syncthreads();
    for (int off = 128; off > 0; off >>= 1) {
        if (tid < off) { rs[tid] += rs[tid + off]; rq[tid] += rq[tid + off]; }
        __syncthreads();
    }
    if (tid == 0) {
        atomicAdd(&sum[bg], rs[0]);
        atomicAdd(&sumsq[bg], rq[0]);
    }
}

__global__ void gn_final(const float* __restrict__ sum, const float* __restrict__ sumsq,
                         float* __restrict__ mean, float* __restrict__ rstd) {
    int i = threadIdx.x;
    if (i < 16) {
        float m = sum[i] / GN_N;
        float v = sumsq[i] / GN_N - m * m;
        mean[i] = m;
        rstd[i] = rsqrtf(v + EPSF);
    }
}

// Normalize + affine; emit K-pair-interleaved bf16:
//   xnP[(b*256 + c/2)*S + s] = bf16(xn[c]) | bf16(xn[c+1])<<16
// (exactly the WMMA B-operand dword format; lets the TDM copy tiles verbatim)
__global__ void gn_apply(const float* __restrict__ x,
                         const float* __restrict__ gamma, const float* __restrict__ beta,
                         const float* __restrict__ mean, const float* __restrict__ rstd,
                         unsigned* __restrict__ xnP) {
    const size_t t  = (size_t)blockIdx.x * blockDim.x + threadIdx.x;   // < 4194304
    const int  s4 = (int)(t & 8191);           // S/4 positions
    const int  kp = (int)((t >> 13) & 255);    // channel pair
    const int  b  = (int)(t >> 21);
    const int  c0 = 2 * kp;
    const int  bg = b * NGRP + (c0 >> 6);

    const size_t base = ((size_t)(b * Cc + c0)) * Ss + (size_t)s4 * 4;
    float4 u = *(const float4*)(x + base);         // channel c0
    float4 v = *(const float4*)(x + base + Ss);    // channel c0+1

    float a0 = gamma[c0] * rstd[bg],     d0 = beta[c0]     - mean[bg] * a0;
    float a1 = gamma[c0 + 1] * rstd[bg], d1 = beta[c0 + 1] - mean[bg] * a1;

    uint4 o;
    o.x = pack2(u.x * a0 + d0, v.x * a1 + d1);
    o.y = pack2(u.y * a0 + d0, v.y * a1 + d1);
    o.z = pack2(u.z * a0 + d0, v.z * a1 + d1);
    o.w = pack2(u.w * a0 + d0, v.w * a1 + d1);
    *(uint4*)&xnP[((size_t)(b * 256 + kp)) * Ss + (size_t)s4 * 4] = o;
}

// Convert both weight matrices to bf16 (row-major (o,c); c-pairs contiguous)
__global__ void wconv(const float* __restrict__ W1, const float* __restrict__ W2,
                      __bf16* __restrict__ W1b, __bf16* __restrict__ W2b) {
    const size_t i = (size_t)blockIdx.x * blockDim.x + threadIdx.x;   // < 65536
    float4 a = ((const float4*)W1)[i];
    float4 b = ((const float4*)W2)[i];
    uint2 oa, ob;
    oa.x = pack2(a.x, a.y); oa.y = pack2(a.z, a.w);
    ob.x = pack2(b.x, b.y); ob.y = pack2(b.z, b.w);
    ((uint2*)W1b)[i] = oa;
    ((uint2*)W2b)[i] = ob;
}

// ---------------------------- TDM descriptors ------------------------------
// D# group0: count=1 | lds_addr | global_addr[56:0] | type=2  (ISA 8.3)
// D# group1: data_size=4B, pad_enable, pad_interval/amount, dims (ISA 8.4)
// This toolchain's builtin takes 6 args: (g0 u32x4, g1 i32x8, g2 i32x4,
// g3 i32x4, i32x8, i32 cpol); groups 2/3 NULL-equivalent (zero) for 2D.
__device__ __forceinline__ void tdm_load(unsigned lds_off, const void* gaddr,
                                         i32x8v g1) {
    unsigned long long ga = (unsigned long long)(size_t)gaddr;
    u32x4v g0;
    g0.x = 1u;                                      // count=1 (valid), user mode
    g0.y = lds_off;                                 // lds_addr (bytes)
    g0.z = (unsigned)ga;                            // global_addr[31:0]
    g0.w = (unsigned)(ga >> 32) | (2u << 30);       // addr[56:32] | type=2
    i32x4v z4 = {0, 0, 0, 0};
    i32x8v z8 = {0, 0, 0, 0, 0, 0, 0, 0};
    __builtin_amdgcn_tensor_load_to_lds(g0, g1, z4, z4, z8, 0);
}

// ------------------------------- bf16 GEMM ---------------------------------
// OUT(o, s) = sum_c A(o, c) * B(c, s)  per batch (blockIdx.z)
// Block: 256 threads = 8 waves; tile 128(M) x 128(N); K-step 32; 16 steps.
// Wave grid 2x4; each wave: 4x2 16x16 tiles (v_wmma_f32_16x16x32_bf16).
// Tiles staged by TDM into double-buffered LDS; wave 0 drives the DMA.
// MODE 0: out = bf16-pair(gelu(acc + bias)) -> outP   (u32, K-pair format)
// MODE 1: out = xres + acc + bias           -> outf   (fp32)
template <int MODE>
__global__ __launch_bounds__(256)
void gemm_kernel(const unsigned* __restrict__ A32,   // weights (o x C/2 u32)
                 const unsigned* __restrict__ B32,   // activations (C/2 x S u32)
                 const float* __restrict__ bias, const float* __restrict__ xres,
                 float* __restrict__ outf, unsigned* __restrict__ outP) {
    __shared__ unsigned smem[2 * BUF_DW];

    const int tid    = threadIdx.x;
    const int wave   = tid >> 5;
    const int lane   = tid & 31;
    const int r      = lane & 15;      // row/col index within 16
    const int h      = lane >> 4;      // half select (K partition)
    const int wave_m = wave >> 2;      // 0..1  (64 rows each)
    const int wave_n = wave & 3;       // 0..3  (32 cols each)

    const int n0 = blockIdx.x * 128;               // spatial tile base
    const int m0 = blockIdx.y * 128;               // out-channel tile base
    const size_t boffP = (size_t)blockIdx.z * 256 * Ss;   // u32 batch offset
    const unsigned* Bbat = B32 + boffP;

    // --- TDM descriptors (group1 static; group0 rebuilt per tile) ---
    // A: u32 tensor 512 rows x 256 cols; tile 128 x 16; LDS stride 20 dw
    const i32x8v g1A = { (int)((2u << 16) | (1u << 20) | (3u << 22) | (3u << 25)),
                         (int)(256u << 16),   // tensor_dim0 = 256 (bits 79:48)
                         (int)(512u << 16),   // tensor_dim1 = 512 (bits 111:80)
                         (int)(16u << 16),    // tile_dim0   = 16  (bits 127:112)
                         16 * 8,              // tile_dim1   = 128 (bits 143:128)
                         256,                 // dim0_stride = 256 (bits 207:160)
                         0, 0 };
    // B: u32 tensor 256 rows x 32768 cols; tile 16 x 128; LDS stride 132 dw
    const i32x8v g1B = { (int)((2u << 16) | (1u << 20) | (6u << 22) | (3u << 25)),
                         (int)(32768u << 16), // tensor_dim0 = 32768 low16<<16
                         (int)(256u << 16),   // tensor_dim1 = 256
                         (int)(128u << 16),   // tile_dim0   = 128
                         16,                  // tile_dim1   = 16
                         32768,               // dim0_stride = 32768
                         0, 0 };
    const unsigned smemBase = (unsigned)(size_t)&smem[0];  // LDS byte offset

    if (wave == 0) {   // prologue: stage tile 0 into buffer 0
        tdm_load(smemBase,              A32 + (size_t)m0 * 256, g1A);
        tdm_load(smemBase + SA_DW * 4,  Bbat + n0,              g1B);
    }

    v8f acc[4][2] = {};

    for (int kt = 0; kt < 16; ++kt) {
        if (wave == 0) {
            if (kt < 15) {   // stage tile kt+1 into the other buffer
                const unsigned nb = smemBase + ((kt + 1) & 1) * (BUF_DW * 4);
                const int kp0 = (kt + 1) * 16;
                tdm_load(nb,             A32 + (size_t)m0 * 256 + kp0, g1A);
                tdm_load(nb + SA_DW * 4, Bbat + (size_t)kp0 * Ss + n0, g1B);
                __builtin_amdgcn_s_wait_tensorcnt(2);   // tile kt complete
            } else {
                __builtin_amdgcn_s_wait_tensorcnt(0);
            }
        }
        __syncthreads();                       // publish buffer kt&1

        const unsigned cb = (kt & 1) * BUF_DW; // dword base of current buffer
        // ---- fragments: contiguous k order (k = 16h + s) in both A and B ----
        // A (16x32 MxK): lane r = row M; h=0 -> K 0..15, h=1 -> K 16..31.
        v16bf afr[4];
#pragma unroll
        for (int mt = 0; mt < 4; ++mt) {
            int row = wave_m * 64 + mt * 16 + r;
            uint4 lo = *(const uint4*)&smem[cb + row * SA_STRIDE + h * 8];
            uint4 hi = *(const uint4*)&smem[cb + row * SA_STRIDE + h * 8 + 4];
            v8u aw = {lo.x, lo.y, lo.z, lo.w, hi.x, hi.y, hi.z, hi.w};
            afr[mt] = __builtin_bit_cast(v16bf, aw);
        }
        // B (32x16 KxN): lane r = col N; h=0 -> K 0..15, h=1 -> K 16..31.
        v16bf bfr[2];
#pragma unroll
        for (int nt = 0; nt < 2; ++nt) {
            int col = wave_n * 32 + nt * 16 + r;
            v8u bw;
#pragma unroll
            for (int j = 0; j < 8; ++j)
                bw[j] = smem[cb + SA_DW + (h * 8 + j) * SB_STRIDE + col];
            bfr[nt] = __builtin_bit_cast(v16bf, bw);
        }
        // ---- 8 WMMAs per K-step ----
#pragma unroll
        for (int mt = 0; mt < 4; ++mt)
#pragma unroll
            for (int nt = 0; nt < 2; ++nt)
                acc[mt][nt] = __builtin_amdgcn_wmma_f32_16x16x32_bf16(
                    false, afr[mt], false, bfr[nt],
                    (short)0, acc[mt][nt], false, false);
        __syncthreads();                       // buffer may be overwritten next
    }

    // ---- epilogue: D layout lane l, vgpr v -> M = 8*(l>>4)+v, N = l&15 ----
#pragma unroll
    for (int mt = 0; mt < 4; ++mt) {
#pragma unroll
        for (int nt = 0; nt < 2; ++nt) {
            int s = n0 + wave_n * 32 + nt * 16 + r;
            int ob = m0 + wave_m * 64 + mt * 16 + h * 8;   // even base row
            if (MODE == 0) {
#pragma unroll
                for (int v = 0; v < 8; v += 2) {           // pack row pairs
                    float v0 = acc[mt][nt][v]     + bias[ob + v];
                    float v1 = acc[mt][nt][v + 1] + bias[ob + v + 1];
                    float g0 = 0.5f * v0 * (1.0f + erff(v0 * 0.70710678118654752f));
                    float g1 = 0.5f * v1 * (1.0f + erff(v1 * 0.70710678118654752f));
                    outP[boffP + ((size_t)((ob + v) >> 1)) * Ss + s] = pack2(g0, g1);
                }
            } else {
#pragma unroll
                for (int v = 0; v < 8; ++v) {
                    int o = ob + v;
                    size_t idx = (size_t)blockIdx.z * Cc * Ss + (size_t)o * Ss + s;
                    outf[idx] = xres[idx] + acc[mt][nt][v] + bias[o];
                }
            }
        }
    }
}

// ------------------------------- launcher ----------------------------------

extern "C" void kernel_launch(void* const* d_in, const int* in_sizes, int n_in,
                              void* d_out, int out_size, void* d_ws, size_t ws_size,
                              hipStream_t stream) {
    const float* x     = (const float*)d_in[0];
    const float* gamma = (const float*)d_in[1];
    const float* beta  = (const float*)d_in[2];
    const float* W1    = (const float*)d_in[3];
    const float* b1    = (const float*)d_in[4];
    const float* W2    = (const float*)d_in[5];
    const float* b2    = (const float*)d_in[6];
    float* out = (float*)d_out;

    char* ws = (char*)d_ws;
    // workspace layout (256-aligned)
    float*    sum   = (float*)(ws + 0);          // 16
    float*    sumsq = (float*)(ws + 64);         // 16
    float*    mean  = (float*)(ws + 128);        // 16
    float*    rstd  = (float*)(ws + 192);        // 16
    __bf16*   W1b   = (__bf16*)(ws + 1024);                 // 512 KiB
    __bf16*   W2b   = (__bf16*)(ws + 1024 + 524288);        // 512 KiB
    unsigned* xnP   = (unsigned*)(ws + 1024 + 2 * 524288);  // 64 MiB (C/2 x S u32)
    unsigned* hP    = (unsigned*)(ws + 1024 + 2 * 524288 + (size_t)67108864); // 64 MiB

    zero_stats<<<1, 32, 0, stream>>>(sum);
    gn_partial<<<2048, 256, 0, stream>>>(x, sum, sumsq);
    gn_final<<<1, 32, 0, stream>>>(sum, sumsq, mean, rstd);
    gn_apply<<<16384, 256, 0, stream>>>(x, gamma, beta, mean, rstd, xnP);
    wconv<<<256, 256, 0, stream>>>(W1, W2, W1b, W2b);

    dim3 ggrid(Ss / 128, Cc / 128, Bb);   // (256, 4, 2)
    gemm_kernel<0><<<ggrid, 256, 0, stream>>>((const unsigned*)W1b, xnP, b1,
                                              nullptr, nullptr, hP);
    gemm_kernel<1><<<ggrid, 256, 0, stream>>>((const unsigned*)W2b, hP, b2,
                                              x, out, nullptr);
}
